// TrainablePQCQCNN_33535104647644
// MI455X (gfx1250) — matrix operand order repaired and checked by reading, not compiled, not run on hardware
//
#include <hip/hip_runtime.h>

#define NQ    6
#define DIM   64
#define NL    3
#define FEAT  512
#define ENCH  32
#define HH    128
#define GT    1296      // 54*24
#define BTOT  32768
#define BLK   128       // threads per block == samples per block
#define WAVES 4
#define GPAD  136       // g row stride in halfs (128 + 8 pad -> 4-bank row stride)

typedef __attribute__((ext_vector_type(16))) _Float16 v16h;
typedef __attribute__((ext_vector_type(8)))  float    v8f;

__global__ __launch_bounds__(BLK) void pqc_fused(
    const float* __restrict__ x,
    const float* __restrict__ enc_w1, const float* __restrict__ enc_b1,
    const float* __restrict__ enc_w2, const float* __restrict__ enc_b2,
    const float* __restrict__ qw,
    const float* __restrict__ head_w1, const float* __restrict__ head_b1,
    const float* __restrict__ head_w2, const float* __restrict__ head_b2,
    float* __restrict__ out)
{
    __shared__ __align__(16) _Float16 sg[BLK * GPAD];   // g, fp16, padded rows
    __shared__ __align__(16) _Float16 sb[16 * GPAD];    // head_w2 tile, fp16, [n][k]

    const int tid = threadIdx.x;
    const int s   = blockIdx.x * BLK + tid;

    // ---------------- Phase 1a: encoder GEMM1 (x row in VMEM, w1 uniform) ----
    float acc[ENCH];
    #pragma unroll
    for (int j = 0; j < ENCH; ++j) acc[j] = 0.f;

    const float4* x4 = (const float4*)(x + (size_t)s * FEAT);
    for (int kk = 0; kk < FEAT / 4; ++kk) {
        float4 xv = x4[kk];
        const float* w = enc_w1 + kk * 4 * ENCH;   // uniform address -> s_load
        #pragma unroll
        for (int j = 0; j < ENCH; ++j) {
            acc[j] = fmaf(xv.x, w[j],            acc[j]);
            acc[j] = fmaf(xv.y, w[ENCH + j],     acc[j]);
            acc[j] = fmaf(xv.z, w[2 * ENCH + j], acc[j]);
            acc[j] = fmaf(xv.w, w[3 * ENCH + j], acc[j]);
        }
    }

    // ---------------- Phase 1b: tanh + GEMM2 -> angles ----------------------
    float ang[NQ];
    #pragma unroll
    for (int q = 0; q < NQ; ++q) ang[q] = enc_b2[q];
    #pragma unroll
    for (int j = 0; j < ENCH; ++j) {
        float h = tanhf(acc[j] + enc_b1[j]);
        #pragma unroll
        for (int q = 0; q < NQ; ++q) ang[q] = fmaf(h, enc_w2[j * NQ + q], ang[q]);
    }

    // ---------------- Phase 1c: 6-qubit state-vector sim (registers) --------
    float sr[DIM], si[DIM];
    #pragma unroll
    for (int i = 0; i < DIM; ++i) { sr[i] = 0.f; si[i] = 0.f; }
    sr[0] = 1.f;

    float cE[NQ], sE[NQ];
    #pragma unroll
    for (int q = 0; q < NQ; ++q) { cE[q] = __cosf(0.5f * ang[q]); sE[q] = __sinf(0.5f * ang[q]); }

    #pragma unroll 1
    for (int layer = 0; layer < NL; ++layer) {
        // data re-uploading: RY(angle[q]) on each qubit (qubit q == bit 5-q)
        #pragma unroll
        for (int q = 0; q < NQ; ++q) {
            const int bit = 1 << (5 - q);
            const float c = cE[q], sv = sE[q];
            #pragma unroll
            for (int i = 0; i < DIM; ++i) {
                if (i & bit) continue;
                const int j = i | bit;
                float ar = sr[i], ai = si[i], br = sr[j], bi = si[j];
                sr[i] = c * ar - sv * br;  si[i] = c * ai - sv * bi;
                sr[j] = sv * ar + c * br;  si[j] = sv * ai + c * bi;
            }
        }
        // trainable Rot = RZ(om) RY(th) RZ(phi)
        #pragma unroll
        for (int q = 0; q < NQ; ++q) {
            const float* w = qw + (layer * NQ + q) * 3;     // uniform -> scalar
            float phi = w[0], th = w[1], om = w[2];
            float ct = __cosf(0.5f * th), st = __sinf(0.5f * th);
            float ap = 0.5f * (phi + om), am = 0.5f * (phi - om);
            float epr = __cosf(ap), epi = -__sinf(ap);
            float emr = __cosf(am), emi = -__sinf(am);
            float u00r =  epr * ct, u00i =  epi * ct;
            float u01r = -emr * st, u01i =  emi * st;
            float u10r =  emr * st, u10i =  emi * st;
            float u11r =  epr * ct, u11i = -epi * ct;
            const int bit = 1 << (5 - q);
            #pragma unroll
            for (int i = 0; i < DIM; ++i) {
                if (i & bit) continue;
                const int j = i | bit;
                float ar = sr[i], ai = si[i], br = sr[j], bi = si[j];
                sr[i] = u00r * ar - u00i * ai + u01r * br - u01i * bi;
                si[i] = u00r * ai + u00i * ar + u01r * bi + u01i * br;
                sr[j] = u10r * ar - u10i * ai + u11r * br - u11i * bi;
                si[j] = u10r * ai + u10i * ar + u11r * bi + u11i * br;
            }
        }
        // CNOT ring: (0,1)(2,3)(4,5)(1,2)(3,4)(5,0) -- pure register renames
        const int ctl[6] = {0, 2, 4, 1, 3, 5};
        const int tgt[6] = {1, 3, 5, 2, 4, 0};
        #pragma unroll
        for (int e = 0; e < 6; ++e) {
            const int bc = 1 << (5 - ctl[e]);
            const int bt = 1 << (5 - tgt[e]);
            #pragma unroll
            for (int i = 0; i < DIM; ++i) {
                if ((i & bc) && !(i & bt)) {
                    const int j = i | bt;
                    float tr = sr[i]; sr[i] = sr[j]; sr[j] = tr;
                    float ti = si[i]; si[i] = si[j]; si[j] = ti;
                }
            }
        }
    }

    // <Z_q> expectation values
    float z[NQ];
    #pragma unroll
    for (int q = 0; q < NQ; ++q) z[q] = 0.f;
    #pragma unroll
    for (int i = 0; i < DIM; ++i) {
        float p = sr[i] * sr[i] + si[i] * si[i];
        #pragma unroll
        for (int q = 0; q < NQ; ++q) z[q] += (i & (1 << (5 - q))) ? -p : p;
    }

    // ---------------- Phase 1d: head GEMM1 + relu -> g (fp16) into LDS ------
    for (int j0 = 0; j0 < HH; j0 += 8) {
        union { float4 f; _Float16 e[8]; } pk;
        #pragma unroll
        for (int u = 0; u < 8; ++u) {
            const int j = j0 + u;
            float v = head_b1[j];
            #pragma unroll
            for (int q = 0; q < NQ; ++q) v = fmaf(z[q], head_w1[q * HH + j], v);
            pk.e[u] = (_Float16)fmaxf(v, 0.f);
        }
        *(float4*)&sg[tid * GPAD + j0] = pk.f;
    }
    __syncthreads();

    // ---------------- Phase 2: head GEMM2 via WMMA f16 ----------------------
    const int lane = tid & 31;
    const int wv   = tid >> 5;
    const int col  = lane & 15;     // N within tile / M row for A
    const int hi   = lane >> 4;     // K-half selector per ISA fragment layout
    const size_t srow0 = (size_t)blockIdx.x * BLK;

    for (int nt = 0; nt < GT / 16; ++nt) {
        const int n0 = nt * 16;
        // stage head_w2[k][n0+n] -> sb[n][k] as fp16 (transposed, K-contiguous)
        #pragma unroll
        for (int i = 0; i < 16; ++i) {
            const int idx = tid + BLK * i;          // 0..2047
            const int k = idx >> 4, n = idx & 15;
            sb[n * GPAD + k] = (_Float16)head_w2[k * GT + n0 + n];
        }
        __syncthreads();

        const float bias = head_b2[n0 + col];

        #pragma unroll
        for (int mtl = 0; mtl < 2; ++mtl) {
            const int mt = wv + mtl * WAVES;        // 0..7 -> 8 M-tiles of 16
            const _Float16* pA = &sg[(mt * 16 + col) * GPAD];
            const _Float16* pB = &sb[col * GPAD];
            v8f c = {0.f, 0.f, 0.f, 0.f, 0.f, 0.f, 0.f, 0.f};
            #pragma unroll
            for (int ks = 0; ks < HH; ks += 32) {
                union { v16h v; float4 f[2]; } a, b;
                // A 16x32 f16: lanes0-15 K[0..7|16..23], lanes16-31 K[8..15|24..31]
                a.f[0] = *(const float4*)(pA + ks + hi * 8);
                a.f[1] = *(const float4*)(pA + ks + hi * 8 + 16);
                // B 32x16 f16: lanes0-15 K[0..7|16..23], lanes16-31 K[8..15|24..31]
                b.f[0] = *(const float4*)(pB + ks + hi * 8);
                b.f[1] = *(const float4*)(pB + ks + hi * 8 + 16);
                c = __builtin_amdgcn_wmma_f32_16x16x32_f16(
                        false, a.v, false, b.v, (short)0, c, false, false);
            }
            // D layout: VGPR i -> M row (lanes16-31: +8), lane%16 -> N
            const size_t base = (srow0 + (size_t)mt * 16) * GT + n0 + col;
            #pragma unroll
            for (int i = 0; i < 8; ++i) {
                const int m = i + hi * 8;
                out[base + (size_t)m * GT] = c[i] + bias;
            }
        }
        __syncthreads();
    }
}

extern "C" void kernel_launch(void* const* d_in, const int* in_sizes, int n_in,
                              void* d_out, int out_size, void* d_ws, size_t ws_size,
                              hipStream_t stream) {
    (void)in_sizes; (void)n_in; (void)d_ws; (void)ws_size; (void)out_size;
    const float* x   = (const float*)d_in[0];
    const float* ew1 = (const float*)d_in[1];
    const float* eb1 = (const float*)d_in[2];
    const float* ew2 = (const float*)d_in[3];
    const float* eb2 = (const float*)d_in[4];
    const float* qw  = (const float*)d_in[5];
    const float* hw1 = (const float*)d_in[6];
    const float* hb1 = (const float*)d_in[7];
    const float* hw2 = (const float*)d_in[8];
    const float* hb2 = (const float*)d_in[9];
    float* outp = (float*)d_out;

    pqc_fused<<<dim3(BTOT / BLK), dim3(BLK), 0, stream>>>(
        x, ew1, eb1, ew2, eb2, qw, hw1, hb1, hw2, hb2, outp);
}
